// Linear_7876970021358
// MI455X (gfx1250) — compile-verified, matrix operand order checked by reading, not run on hardware
//
#include <hip/hip_runtime.h>

typedef __attribute__((ext_vector_type(16))) __bf16 v16bf;
typedef __attribute__((ext_vector_type(8)))  __bf16 v8bf;
typedef __attribute__((ext_vector_type(8)))  float  v8f;

#define MULC    512
#define NROW    8192
#define TOTALC  8192
#define KCHUNK  128
#define ROWS    32          // rows per block (2 row-tiles of 16)
#define ASTRIDE 136         // bf16 LDS row stride (68 dwords; 68%64=4 -> near-conflict-free b128)
#define SEG_WS  (16 * 32 * 1024)  // per-segment swizzled W elements: 16 kc x 32 ct x (2 planes x 32 lanes x 16)

// ---------------------------------------------------------------------------
// Pre-kernel: convert f32 W -> (hi, lo) bf16 planes, pre-swizzled into the
// exact per-lane V_WMMA_F32_16X16X32_BF16 B-fragment order, so the GEMM kernel
// reads each fragment as one contiguous 32B per lane (2x global_load_b128).
// Layout: [seg][kc][ct][plane(hi=0,lo=1)][lane][16]  (element index == tid).
// ISA 16-bit B 32x16: lanes 0-15 -> N=lane, K {0..7,16..23};
//                     lanes 16-31 -> N=lane-16, K {8..15,24..31}.
// ---------------------------------------------------------------------------
__global__ __launch_bounds__(256) void pack_w_bf16x2(
    const float* __restrict__ w0, const float* __restrict__ w1,
    const float* __restrict__ w2, const float* __restrict__ w3,
    __bf16* __restrict__ wsbf)
{
    const unsigned t = blockIdx.x * 256u + threadIdx.x;   // 2,097,152 total
    const int i     = t & 15;
    const int lane  = (t >> 4) & 31;
    const int plane = (t >> 9) & 1;
    const int ct    = (t >> 10) & 31;
    const int kc    = (t >> 15) & 15;
    const int s     = t >> 19;

    const float* w = (s == 0) ? w0 : (s == 1) ? w1 : (s == 2) ? w2 : w3;

    const int v     = ct * 16 + (lane & 15);
    const int kbase = kc * 32 + ((lane >> 4) ? 8 : 0);
    const int k     = kbase + ((i < 8) ? i : (8 + i));    // {0..7} then {+16..+23}

    const float f = w[(size_t)k * MULC + v];
    const __bf16 hi = (__bf16)f;
    wsbf[t] = (plane == 0) ? hi : (__bf16)(f - (float)hi);
}

// ---------------------------------------------------------------------------
// Main kernel: block-diagonal equivariant linear as 16 f32 GEMMs
// (M=8192, K=512, N=512) in bf16_3x (Precision.HIGH) on
// V_WMMA_F32_16X16X32_BF16: acc += Ah*Bh + Ah*Bl + Al*Bh, f32 accumulate.
// Block = 256 threads (8 wave32): 32 rows x 512 cols; wave = 2 row-tiles x 4 col-tiles.
// ---------------------------------------------------------------------------
__global__ __launch_bounds__(256) void eqlin_wmma_bf16x3(
    const float* __restrict__ x,
    const __bf16* __restrict__ wsbf,
    float* __restrict__ out)
{
    __shared__ __align__(16) __bf16 lds_h[ROWS][ASTRIDE];
    __shared__ __align__(16) __bf16 lds_l[ROWS][ASTRIDE];

    // Decode (segment, component): dims 1,3,5,7 -> 16 components.
    const int comp = blockIdx.y;
    int s, d, c, off;
    if (comp < 1)      { s = 0; d = 1; c = comp;     off = 0;    }
    else if (comp < 4) { s = 1; d = 3; c = comp - 1; off = 512;  }
    else if (comp < 9) { s = 2; d = 5; c = comp - 4; off = 2048; }
    else               { s = 3; d = 7; c = comp - 9; off = 4608; }

    const int row0  = blockIdx.x * ROWS;
    const int tid   = threadIdx.x;
    const int lane  = tid & 31;
    const int wave  = tid >> 5;        // 0..7
    const int mrow  = lane & 15;
    const int ksel  = (lane >> 4) * 8; // 16-bit A/B layout: K sub-block select
    const int vbase = wave * 64;       // this wave's 64 output columns

    v8f acc[2][4] = {};                // [row-tile][col-tile]

    // Cooperative X staging: thread t loads row t>>3 (0..31), k = (t&7) + 8*i.
    const int srow = tid >> 3;
    const int skk  = tid & 7;
    const float* __restrict__ xrow =
        x + (size_t)(row0 + srow) * TOTALC + off + c;

    // Per-lane A-fragment bases (hi/lo planes, both row-tiles).
    const __bf16* __restrict__ ah0 = &lds_h[mrow][ksel];
    const __bf16* __restrict__ ah1 = &lds_h[16 + mrow][ksel];
    const __bf16* __restrict__ al0 = &lds_l[mrow][ksel];
    const __bf16* __restrict__ al1 = &lds_l[16 + mrow][ksel];

    // Per-lane B-fragment base: this wave's first col-tile, this lane's 16 elems.
    const __bf16* __restrict__ bfrag0 =
        wsbf + (size_t)s * SEG_WS + (size_t)(wave * 4) * 1024 + lane * 16;

    for (int k0 = 0; k0 < MULC; k0 += KCHUNK) {
        // Stage X chunk with on-the-fly f32 -> bf16 hi/lo split.
        #pragma unroll
        for (int i = 0; i < KCHUNK / 8; ++i) {
            const int k = skk + i * 8;
            const float f = xrow[(size_t)(k0 + k) * d];
            const __bf16 h = (__bf16)f;
            lds_h[srow][k] = h;
            lds_l[srow][k] = (__bf16)(f - (float)h);
        }
        if (k0 + KCHUNK < MULC) {
            __builtin_prefetch(xrow + (size_t)(k0 + KCHUNK + skk * 16) * d, 0, 0);
        }
        __syncthreads();

        #pragma unroll
        for (int kc = 0; kc < KCHUNK / 32; ++kc) {
            // A fragments: two contiguous 8-element halves per ISA 16-bit layout.
            const int ko = kc * 32;
            v16bf Ah0 = __builtin_shufflevector(
                *(const v8bf*)(ah0 + ko), *(const v8bf*)(ah0 + ko + 16),
                0,1,2,3,4,5,6,7,8,9,10,11,12,13,14,15);
            v16bf Ah1 = __builtin_shufflevector(
                *(const v8bf*)(ah1 + ko), *(const v8bf*)(ah1 + ko + 16),
                0,1,2,3,4,5,6,7,8,9,10,11,12,13,14,15);
            v16bf Al0 = __builtin_shufflevector(
                *(const v8bf*)(al0 + ko), *(const v8bf*)(al0 + ko + 16),
                0,1,2,3,4,5,6,7,8,9,10,11,12,13,14,15);
            v16bf Al1 = __builtin_shufflevector(
                *(const v8bf*)(al1 + ko), *(const v8bf*)(al1 + ko + 16),
                0,1,2,3,4,5,6,7,8,9,10,11,12,13,14,15);

            const int kcg = (k0 >> 5) + kc;
            #pragma unroll
            for (int t = 0; t < 4; ++t) {
                const __bf16* bp = bfrag0 + (size_t)(kcg * 32 + t) * 1024;
                v16bf Bh = *(const v16bf*)bp;          // hi plane (2x b128)
                v16bf Bl = *(const v16bf*)(bp + 512);  // lo plane (2x b128)
                // bf16_3x: hi*hi + hi*lo + lo*hi, f32 accumulate.
                acc[0][t] = __builtin_amdgcn_wmma_f32_16x16x32_bf16(
                    false, Ah0, false, Bh, (short)0, acc[0][t], false, false);
                acc[1][t] = __builtin_amdgcn_wmma_f32_16x16x32_bf16(
                    false, Ah1, false, Bh, (short)0, acc[1][t], false, false);
                acc[0][t] = __builtin_amdgcn_wmma_f32_16x16x32_bf16(
                    false, Ah0, false, Bl, (short)0, acc[0][t], false, false);
                acc[1][t] = __builtin_amdgcn_wmma_f32_16x16x32_bf16(
                    false, Ah1, false, Bl, (short)0, acc[1][t], false, false);
                acc[0][t] = __builtin_amdgcn_wmma_f32_16x16x32_bf16(
                    false, Al0, false, Bh, (short)0, acc[0][t], false, false);
                acc[1][t] = __builtin_amdgcn_wmma_f32_16x16x32_bf16(
                    false, Al1, false, Bh, (short)0, acc[1][t], false, false);
            }
        }
        __syncthreads();
    }

    // D layout: VGPR j -> (M = j + 8*(lane>>4), N = mrow). Scale by 1/sqrt(512).
    const float scale = 0.04419417382415922f;
    #pragma unroll
    for (int rt = 0; rt < 2; ++rt) {
        const int mbase = row0 + rt * 16 + (lane >> 4) * 8;
        #pragma unroll
        for (int t = 0; t < 4; ++t) {
            float* __restrict__ op =
                out + off + (size_t)(vbase + t * 16 + mrow) * d + c;
            #pragma unroll
            for (int j = 0; j < 8; ++j) {
                op[(size_t)(mbase + j) * TOTALC] = acc[rt][t][j] * scale;
            }
        }
    }
}

extern "C" void kernel_launch(void* const* d_in, const int* in_sizes, int n_in,
                              void* d_out, int out_size, void* d_ws, size_t ws_size,
                              hipStream_t stream) {
    const float* x  = (const float*)d_in[0];
    const float* w0 = (const float*)d_in[1];
    const float* w1 = (const float*)d_in[2];
    const float* w2 = (const float*)d_in[3];
    const float* w3 = (const float*)d_in[4];
    float*  out  = (float*)d_out;
    __bf16* wsbf = (__bf16*)d_ws;      // 4 segs x 1 MB = 4 MB swizzled hi/lo W

    // Pack W: 4 * SEG_WS = 2,097,152 elements, one per thread.
    pack_w_bf16x2<<<dim3(4 * SEG_WS / 256), dim3(256), 0, stream>>>(
        w0, w1, w2, w3, wsbf);

    // GEMMs: 256 row-blocks x 16 irrep components, 8 wave32 per block.
    eqlin_wmma_bf16x3<<<dim3(NROW / ROWS, 16), dim3(256), 0, stream>>>(
        x, wsbf, out);
}
